// Transformer_63737314673059
// MI455X (gfx1250) — matrix operand order repaired and checked by reading, not compiled, not run on hardware
//
#include <hip/hip_runtime.h>
#include <hip/hip_bf16.h>
#include <math.h>

// ---------------------------------------------------------------------------
// Transformer forward (4 layers) on gfx1250: fp32 in/out, bf16 WMMA compute.
// DIM=384, HEADS=6, DIM_HEAD=64, INNER=384, MLP=1536, B=4, N=2048 -> M=8192.
//
// Strategy: weights converted+transposed to bf16 [N][K] once; activations
// carried in bf16 between matmuls (fp32 kept only where residuals need it).
// GEMM loads B fragments straight from global (L2-resident), A via LDS.
// ---------------------------------------------------------------------------

typedef __attribute__((ext_vector_type(16))) __bf16 v16bf;
typedef __attribute__((ext_vector_type(8)))  float  v8f;
typedef unsigned short ushort_t;

#define M_TOK   8192
#define DIMD    384
#define NQKV    1152
#define NMLP    1536
#define SEQ     2048
#define NHEAD   6
#define DHEAD   64

// ---- helpers ---------------------------------------------------------------

__device__ __forceinline__ ushort_t f2bf(float f) {
    unsigned int u = __float_as_uint(f);
    unsigned int r = u + 0x7FFFu + ((u >> 16) & 1u);   // round-to-nearest-even
    return (ushort_t)(r >> 16);
}

// A fragment 16x32 bf16 from row-major [row][ldk] (LDS or global):
//  lanes 0-15 : VGPR0-3 = K 0..7,  VGPR4-7 = K 16..23
//  lanes16-31 : VGPR0-3 = K 8..15, VGPR4-7 = K 24..31
__device__ __forceinline__ v16bf load_frag_a(const ushort_t* base,
                                             int row0, int koff, int ldk, int lane) {
    int r = row0 + (lane & 15);
    int half = lane >> 4;
    const uint4* p0 = reinterpret_cast<const uint4*>(base + (size_t)r * ldk + koff + half * 8);
    const uint4* p1 = reinterpret_cast<const uint4*>(base + (size_t)r * ldk + koff + 16 + half * 8);
    union { uint4 q[2]; v16bf v; } u;
    u.q[0] = *p0;
    u.q[1] = *p1;
    return u.v;
}

// B fragment 32x16 bf16 from K-contiguous [col][ldk] (LDS or global):
//  lane -> column (lane&15); lanes 0-15 hold K 0..7,16..23, lanes 16-31 K 8..15,24..31
__device__ __forceinline__ v16bf load_frag_b(const ushort_t* baseT,
                                             int col0, int koff, int ldk, int lane) {
    int c = col0 + (lane & 15);
    int half = lane >> 4;
    const uint4* p0 = reinterpret_cast<const uint4*>(baseT + (size_t)c * ldk + koff + half * 8);
    const uint4* p1 = reinterpret_cast<const uint4*>(baseT + (size_t)c * ldk + koff + 16 + half * 8);
    union { uint4 q[2]; v16bf v; } u;
    u.q[0] = *p0;
    u.q[1] = *p1;
    return u.v;
}

__device__ __forceinline__ v8f wmma_bf16(v16bf a, v16bf b, v8f c) {
    return __builtin_amdgcn_wmma_f32_16x16x32_bf16(
        false, a, false, b, (short)0, c, false, false);
}

// ---------------------------------------------------------------------------
// One-time converters
// ---------------------------------------------------------------------------
__global__ __launch_bounds__(256)
void convert_transpose_kernel(const float* __restrict__ W, ushort_t* __restrict__ Wt,
                              int K, int N) {
    int idx = blockIdx.x * 256 + threadIdx.x;   // coalesced read of W[k][n]
    int k = idx / N;
    int n = idx - k * N;
    Wt[(size_t)n * K + k] = f2bf(W[idx]);
}

__global__ __launch_bounds__(256)
void convert_kernel(const float* __restrict__ src, ushort_t* __restrict__ dst) {
    int idx = blockIdx.x * 256 + threadIdx.x;
    dst[idx] = f2bf(src[idx]);
}

// ---------------------------------------------------------------------------
// GEMM: C[M,N] = A[M,K](bf16) * Bt[N,K](bf16)  (+bias)(+GELU)(+residual fp32)
// OUTMODE: 0 = fp32 only, 1 = bf16 only, 2 = both.
// Block tile 128x128x64, 256 threads = 8 wave32s, wave tile 64x32.
// A staged in LDS (16KB); B fragments loaded directly from global (L2 hits).
// ---------------------------------------------------------------------------
template <int OUTMODE, bool BIAS, bool GELU, bool RES>
__global__ __launch_bounds__(256)
void gemm_bf16_kernel(const ushort_t* __restrict__ A, const ushort_t* __restrict__ Bt,
                      float* __restrict__ Cf, ushort_t* __restrict__ Cb,
                      const float* __restrict__ bias, const float* __restrict__ res,
                      int M, int N, int K) {
    __shared__ ushort_t sA[128 * 64];   // [m][k] bf16, BK=64

    const int tid  = threadIdx.x;
    const int lane = tid & 31;
    const int wave = tid >> 5;
    const int wm   = (wave & 1) * 64;
    const int wn   = (wave >> 1) * 32;
    const int bm   = blockIdx.y * 128;
    const int bn   = blockIdx.x * 128;

    v8f acc[4][2];
    for (int mi = 0; mi < 4; ++mi)
        for (int ni = 0; ni < 2; ++ni)
            for (int j = 0; j < 8; ++j) acc[mi][ni][j] = 0.0f;

    const int sm  = tid & 127;          // staging row
    const int sks = (tid >> 7) * 32;    // k sub-offset 0 or 32

    for (int k0 = 0; k0 < K; k0 += 64) {
        // ---- stage A tile: 64B contiguous bf16 per thread -> LDS ----
        {
            const uint4* g = reinterpret_cast<const uint4*>(
                A + (size_t)(bm + sm) * K + k0 + sks);
            uint4 t0 = g[0], t1 = g[1], t2 = g[2], t3 = g[3];
            uint4* d = reinterpret_cast<uint4*>(&sA[sm * 64 + sks]);
            d[0] = t0; d[1] = t1; d[2] = t2; d[3] = t3;
            if (k0 + 64 < K)
                __builtin_prefetch(A + (size_t)(bm + sm) * K + k0 + 64 + sks, 0, 0);
        }
        __syncthreads();

        for (int s = 0; s < 2; ++s) {
            const int ko = s * 32;
            v16bf bf0 = load_frag_b(Bt, bn + wn,      k0 + ko, K, lane);
            v16bf bf1 = load_frag_b(Bt, bn + wn + 16, k0 + ko, K, lane);
            for (int mi = 0; mi < 4; ++mi) {
                v16bf af = load_frag_a(sA, wm + mi * 16, ko, 64, lane);
                acc[mi][0] = wmma_bf16(af, bf0, acc[mi][0]);
                acc[mi][1] = wmma_bf16(af, bf1, acc[mi][1]);
            }
        }
        __syncthreads();
    }

    // ---- epilogue: C/D layout -> VGPR j holds row j (+8 for upper lanes) ----
    const int cn = lane & 15;
    const int rb = (lane >> 4) * 8;
    for (int mi = 0; mi < 4; ++mi) {
        for (int ni = 0; ni < 2; ++ni) {
            const int col = bn + wn + ni * 16 + cn;
            float bv = 0.0f;
            if (BIAS) bv = bias[col];
            for (int j = 0; j < 8; ++j) {
                const int row = bm + wm + mi * 16 + rb + j;
                float v = acc[mi][ni][j] + bv;
                if (GELU) v = 0.5f * v * (1.0f + erff(v * 0.70710678118654752f));
                if (RES)  v += res[(size_t)row * N + col];
                if (OUTMODE != 1) Cf[(size_t)row * N + col] = v;
                if (OUTMODE != 0) Cb[(size_t)row * N + col] = f2bf(v);
            }
        }
    }
}

// ---------------------------------------------------------------------------
// Flash attention: block = 64 query rows of one (batch, head), bf16 qkv in,
// bf16 out. grid = (SEQ/64, B*NHEAD), 256 threads = 8 waves.
// Softmax scale (1/8) folded into the softmax pass.
// ---------------------------------------------------------------------------
__global__ __launch_bounds__(256)
void attn_kernel(const ushort_t* __restrict__ qkv, ushort_t* __restrict__ outp) {
    const int tid  = threadIdx.x;
    const int lane = tid & 31;
    const int wave = tid >> 5;
    const int qb   = blockIdx.x;
    const int bh   = blockIdx.y;
    const int b    = bh / NHEAD;
    const int h    = bh % NHEAD;
    const size_t rowbase = (size_t)b * SEQ;
    const int qcol = h * DHEAD;
    const int kcol = 384 + h * DHEAD;
    const int vcol = 768 + h * DHEAD;
    const float scl = 0.125f;               // DIM_HEAD^-0.5

    __shared__ ushort_t Qs[64 * 64];  // [q][d]
    __shared__ ushort_t Ks[64 * 64];  // [key][d]
    __shared__ ushort_t Vt[64 * 64];  // [d][key]
    __shared__ float    Ss[64 * 64];  // raw scores fp32
    __shared__ ushort_t Ps[64 * 64];  // probs bf16
    __shared__ float mrow[64], lrow[64], srow[64];

    // ---- stage Q: 32B contiguous bf16 per thread ----
    {
        const int r  = tid >> 2;
        const int dp = (tid & 3) * 16;
        const uint4* g = reinterpret_cast<const uint4*>(
            qkv + (rowbase + (size_t)qb * 64 + r) * NQKV + qcol + dp);
        uint4* d = reinterpret_cast<uint4*>(&Qs[r * 64 + dp]);
        d[0] = g[0]; d[1] = g[1];
    }
    if (tid < 64) { mrow[tid] = -3.0e38f; lrow[tid] = 0.0f; }

    v8f oacc[2];
    for (int i = 0; i < 2; ++i)
        for (int j = 0; j < 8; ++j) oacc[i][j] = 0.0f;

    const int t0i = wave * 2;

    for (int kb = 0; kb < SEQ / 64; ++kb) {
        // ---- stage K and V^T ----
        {
            const int r  = tid >> 2;
            const int dp = (tid & 3) * 16;
            const size_t grow = rowbase + (size_t)kb * 64 + r;
            const uint4* gk = reinterpret_cast<const uint4*>(
                qkv + grow * NQKV + kcol + dp);
            uint4* dk = reinterpret_cast<uint4*>(&Ks[r * 64 + dp]);
            dk[0] = gk[0]; dk[1] = gk[1];

            const uint4* gv = reinterpret_cast<const uint4*>(
                qkv + grow * NQKV + vcol + dp);
            union { uint4 q[2]; ushort_t s[16]; } uv;
            uv.q[0] = gv[0]; uv.q[1] = gv[1];
            for (int j = 0; j < 16; ++j)
                Vt[(dp + j) * 64 + r] = uv.s[j];
        }
        __syncthreads();

        // ---- S = Q·K^T (raw) for this wave's two tiles -> Ss ----
        for (int i = 0; i < 2; ++i) {
            const int t  = t0i + i;
            const int mt = (t >> 2) * 16;
            const int nt = (t & 3) * 16;
            v8f s;
            for (int j = 0; j < 8; ++j) s[j] = 0.0f;
            for (int ks = 0; ks < 2; ++ks) {
                v16bf a  = load_frag_a(Qs, mt, ks * 32, 64, lane);
                v16bf bb = load_frag_b(Ks, nt, ks * 32, 64, lane);
                s = wmma_bf16(a, bb, s);
            }
            const int cn = lane & 15;
            const int rb = (lane >> 4) * 8;
            for (int j = 0; j < 8; ++j)
                Ss[(mt + rb + j) * 64 + nt + cn] = s[j];
        }
        __syncthreads();

        // ---- online softmax (scale folded here): one thread per row ----
        if (tid < 64) {
            const float* sr = &Ss[tid * 64];
            float mx = mrow[tid];
            for (int c = 0; c < 64; ++c) mx = fmaxf(mx, sr[c] * scl);
            const float sc = __expf(mrow[tid] - mx);
            float sum = 0.0f;
            for (int c = 0; c < 64; ++c) {
                float p = __expf(sr[c] * scl - mx);
                sum += p;
                Ps[tid * 64 + c] = f2bf(p);
            }
            lrow[tid] = lrow[tid] * sc + sum;
            mrow[tid] = mx;
            srow[tid] = sc;
        }
        __syncthreads();

        // ---- O = O*rescale + P·V ----
        for (int i = 0; i < 2; ++i) {
            const int t  = t0i + i;
            const int mt = (t >> 2) * 16;
            const int nt = (t & 3) * 16;
            const int rb = (lane >> 4) * 8;
            for (int j = 0; j < 8; ++j) oacc[i][j] *= srow[mt + rb + j];
            for (int ks = 0; ks < 2; ++ks) {
                v16bf a  = load_frag_a(Ps, mt, ks * 32, 64, lane);
                v16bf bb = load_frag_b(Vt, nt, ks * 32, 64, lane);
                oacc[i] = wmma_bf16(a, bb, oacc[i]);
            }
        }
        __syncthreads();
    }

    // ---- normalize, write bf16 out: [b*SEQ + q, h*64 + d] into [M,384] ----
    for (int i = 0; i < 2; ++i) {
        const int t  = t0i + i;
        const int mt = (t >> 2) * 16;
        const int nt = (t & 3) * 16;
        const int cn = lane & 15;
        const int rb = (lane >> 4) * 8;
        for (int j = 0; j < 8; ++j) {
            const int r = mt + rb + j;
            const float invl = 1.0f / lrow[r];
            outp[(rowbase + (size_t)qb * 64 + r) * DIMD + h * DHEAD + nt + cn] =
                f2bf(oacc[i][j] * invl);
        }
    }
}

// ---------------------------------------------------------------------------
// Host-side orchestration
// ---------------------------------------------------------------------------
extern "C" void kernel_launch(void* const* d_in, const int* in_sizes, int n_in,
                              void* d_out, int out_size, void* d_ws, size_t ws_size,
                              hipStream_t stream) {
    (void)in_sizes; (void)n_in; (void)out_size; (void)ws_size;

    const float* x    = (const float*)d_in[0];
    const float* Wqkv = (const float*)d_in[1];
    const float* Wout = (const float*)d_in[2];
    const float* W1   = (const float*)d_in[3];
    const float* b1   = (const float*)d_in[4];
    const float* W2   = (const float*)d_in[5];
    const float* b2   = (const float*)d_in[6];
    float* out = (float*)d_out;

    // ---- workspace carve (256B-aligned slabs) ----
    char* p = (char*)d_ws;
    auto carve = [&](size_t bytes) -> char* {
        char* r = p;
        p += (bytes + 255) & ~(size_t)255;
        return r;
    };
    ushort_t* WqkvT = (ushort_t*)carve((size_t)4 * DIMD * NQKV * 2);  // [1152][384] x4
    ushort_t* WoutT = (ushort_t*)carve((size_t)4 * DIMD * DIMD * 2);  // [384][384]  x4
    ushort_t* W1T   = (ushort_t*)carve((size_t)4 * DIMD * NMLP * 2);  // [1536][384] x4
    ushort_t* W2T   = (ushort_t*)carve((size_t)4 * NMLP * DIMD * 2);  // [384][1536] x4
    ushort_t* xbf   = (ushort_t*)carve((size_t)M_TOK * DIMD * 2);
    ushort_t* qkvb  = (ushort_t*)carve((size_t)M_TOK * NQKV * 2);
    ushort_t* attnb = (ushort_t*)carve((size_t)M_TOK * DIMD * 2);
    float*    x1f   = (float*)   carve((size_t)M_TOK * DIMD * 4);
    ushort_t* x1b   = (ushort_t*)carve((size_t)M_TOK * DIMD * 2);
    ushort_t* hb    = (ushort_t*)carve((size_t)M_TOK * NMLP * 2);
    float*    xf    = (float*)   carve((size_t)M_TOK * DIMD * 4);

    const dim3 blk(256);

    // ---- one-time weight convert+transpose, activation convert ----
    for (int l = 0; l < 4; ++l) {
        convert_transpose_kernel<<<DIMD * NQKV / 256, blk, 0, stream>>>(
            Wqkv + (size_t)l * DIMD * NQKV, WqkvT + (size_t)l * DIMD * NQKV, DIMD, NQKV);
        convert_transpose_kernel<<<DIMD * DIMD / 256, blk, 0, stream>>>(
            Wout + (size_t)l * DIMD * DIMD, WoutT + (size_t)l * DIMD * DIMD, DIMD, DIMD);
        convert_transpose_kernel<<<DIMD * NMLP / 256, blk, 0, stream>>>(
            W1 + (size_t)l * DIMD * NMLP, W1T + (size_t)l * DIMD * NMLP, DIMD, NMLP);
        convert_transpose_kernel<<<NMLP * DIMD / 256, blk, 0, stream>>>(
            W2 + (size_t)l * NMLP * DIMD, W2T + (size_t)l * NMLP * DIMD, NMLP, DIMD);
    }
    convert_kernel<<<M_TOK * DIMD / 256, blk, 0, stream>>>(x, xbf);

    const dim3 g_qkv(NQKV / 128, M_TOK / 128);
    const dim3 g_out(DIMD / 128, M_TOK / 128);
    const dim3 g_ff1(NMLP / 128, M_TOK / 128);
    const dim3 g_atn(SEQ / 64, 4 * NHEAD);

    const float* cur_f = x;     // fp32 residual source
    // xbf holds bf16 of current x
    for (int l = 0; l < 4; ++l) {
        const ushort_t* WqkvT_l = WqkvT + (size_t)l * DIMD * NQKV;
        const ushort_t* WoutT_l = WoutT + (size_t)l * DIMD * DIMD;
        const ushort_t* W1T_l   = W1T   + (size_t)l * DIMD * NMLP;
        const ushort_t* W2T_l   = W2T   + (size_t)l * NMLP * DIMD;
        const float*    b1_l    = b1 + (size_t)l * NMLP;
        const float*    b2_l    = b2 + (size_t)l * DIMD;

        // qkv(bf16) = x(bf16) @ Wqkv
        gemm_bf16_kernel<1, false, false, false><<<g_qkv, blk, 0, stream>>>(
            xbf, WqkvT_l, nullptr, qkvb, nullptr, nullptr, M_TOK, NQKV, DIMD);

        // attn(bf16) = softmax(QK^T/8) V
        attn_kernel<<<g_atn, blk, 0, stream>>>(qkvb, attnb);

        // x1 = attn @ Wout + x   (fp32 + bf16)
        gemm_bf16_kernel<2, false, false, true><<<g_out, blk, 0, stream>>>(
            attnb, WoutT_l, x1f, x1b, nullptr, cur_f, M_TOK, DIMD, DIMD);

        // h(bf16) = gelu(x1 @ W1 + b1)
        gemm_bf16_kernel<1, true, true, false><<<g_ff1, blk, 0, stream>>>(
            x1b, W1T_l, nullptr, hb, b1_l, nullptr, M_TOK, NMLP, DIMD);

        // x' = h @ W2 + b2 + x1   (fp32 + bf16; fp32 to d_out on last layer)
        float* dstf = (l == 3) ? out : xf;
        gemm_bf16_kernel<2, true, false, true><<<g_out, blk, 0, stream>>>(
            hb, W2T_l, dstf, xbf, b2_l, x1f, M_TOK, DIMD, NMLP);

        cur_f = dstf;
    }
}
